// GATForNode_47175920779581
// MI455X (gfx1250) — matrix operand order repaired and checked by reading, not compiled, not run on hardware
//
#include <hip/hip_runtime.h>
#include <hip/hip_bf16.h>

typedef __bf16 bf16;
typedef __attribute__((ext_vector_type(16))) __bf16 v16bf;
typedef __attribute__((ext_vector_type(8)))  float  v8f;

#define NEG_SLOPE 0.2f

// ---------- monotonic float <-> uint ordering (for atomic segment-max) ----------
__device__ __forceinline__ unsigned f2key(float f) {
  unsigned u = __float_as_uint(f);
  return (u & 0x80000000u) ? ~u : (u | 0x80000000u);
}
__device__ __forceinline__ float key2f(unsigned k) {
  unsigned u = (k & 0x80000000u) ? (k & 0x7FFFFFFFu) : ~k;
  return __uint_as_float(u);
}

// ---------- prep: W [K x Nout] f32  ->  Wt [NoutPad x K] bf16 (transposed) ----------
__global__ void prep_w(const float* __restrict__ W, bf16* __restrict__ Wt,
                       int K, int Nout, int NoutPad) {
  int id = blockIdx.x * blockDim.x + threadIdx.x;
  if (id >= NoutPad * K) return;
  int n = id / K, k = id - n * K;
  Wt[id] = (n < Nout) ? (bf16)W[(size_t)k * Nout + n] : (bf16)0.0f;
}

// ---------- WMMA GEMM: C[M x Ncols] = A[M x K](f32, cvt->bf16) @ Bt^T ----------
// Bt is [>=COLTILES*16 x K] bf16 row-major (i.e. B transposed). One wave per
// 16-row tile, COLTILES 16-col accumulators, K stepped by 32.
template<int COLTILES>
__global__ void wmma_gemm(const float* __restrict__ A, const bf16* __restrict__ Bt,
                          float* __restrict__ C, int M, int K, int Ncols) {
  int wave = threadIdx.x >> 5;
  int lane = threadIdx.x & 31;
  int tile = blockIdx.x * (blockDim.x >> 5) + wave;
  if (tile * 16 >= M) return;
  int mrow = tile * 16;
  int mloc = lane & 15;
  int half = lane >> 4;

  const float* arow = A + (size_t)(mrow + mloc) * K;
  v8f acc[COLTILES] = {};

  for (int k0 = 0; k0 < K; k0 += 32) {
    // A fragment: lane holds 16 contiguous K at k0+16*half (matches B)
    const float4* g0 = (const float4*)(arow + k0 + 16 * half);
    const float4* g1 = (const float4*)(arow + k0 + 16 * half + 8);
    float4 f0 = g0[0], f1 = g0[1], f2 = g1[0], f3 = g1[1];
    v16bf a;
    a[0]=(bf16)f0.x;  a[1]=(bf16)f0.y;  a[2]=(bf16)f0.z;  a[3]=(bf16)f0.w;
    a[4]=(bf16)f1.x;  a[5]=(bf16)f1.y;  a[6]=(bf16)f1.z;  a[7]=(bf16)f1.w;
    a[8]=(bf16)f2.x;  a[9]=(bf16)f2.y;  a[10]=(bf16)f2.z; a[11]=(bf16)f2.w;
    a[12]=(bf16)f3.x; a[13]=(bf16)f3.y; a[14]=(bf16)f3.z; a[15]=(bf16)f3.w;

#pragma unroll
    for (int ct = 0; ct < COLTILES; ++ct) {
      // B fragment: lane holds column (ct*16 + mloc), 16 contiguous K at k0+16*half
      const v16bf b = *(const v16bf*)(Bt + (size_t)(ct * 16 + mloc) * K + k0 + 16 * half);
      acc[ct] = __builtin_amdgcn_wmma_f32_16x16x32_bf16(
          false, a, false, b, (short)0, acc[ct], false, false);
    }
  }

#pragma unroll
  for (int ct = 0; ct < COLTILES; ++ct) {
#pragma unroll
    for (int r = 0; r < 8; ++r) {
      int row = mrow + 8 * half + r;       // D layout: VGPR r -> M = r + 8*half
      int col = ct * 16 + mloc;            // lane&15 -> N
      if (col < Ncols) C[(size_t)row * Ncols + col] = acc[ct][r];
    }
  }
}

// ---------- attention scalars: asrc[n,h] = <h[n,h,:], a_src[h,:]> ----------
template<int H, int C>
__global__ void att_scores(const float* __restrict__ h, const float* __restrict__ a_src,
                           const float* __restrict__ a_dst, float* __restrict__ asrc,
                           float* __restrict__ adst, int N) {
  long long id = (long long)blockIdx.x * blockDim.x + threadIdx.x;
  if (id >= (long long)N * H) return;
  int hh = (int)(id % H);
  const float* hp = h + id * C;   // id = n*H + h -> h + (n*H+h)*C
  float s = 0.f, d = 0.f;
#pragma unroll
  for (int c = 0; c < C; ++c) {
    float v = hp[c];
    s += v * a_src[hh * C + c];
    d += v * a_dst[hh * C + c];
  }
  asrc[id] = s;
  adst[id] = d;
}

// ---------- edge pass A: segment max of leaky_relu(asrc[src]+adst[dst]) ----------
template<int H>
__global__ void edge_max_k(const long long* __restrict__ ei, long long Eg, long long E,
                           const float* __restrict__ asrc, const float* __restrict__ adst,
                           unsigned* __restrict__ m) {
  long long id = (long long)blockIdx.x * blockDim.x + threadIdx.x;
  if (id >= Eg * H) return;
  long long e = id / H;
  int hh = (int)(id - e * H);
  int src, dst;
  if (e < E) { src = (int)ei[e]; dst = (int)ei[E + e]; }
  else       { src = dst = (int)(e - E); }          // self loop
  float v = asrc[(size_t)src * H + hh] + adst[(size_t)dst * H + hh];
  v = v > 0.f ? v : NEG_SLOPE * v;
  atomicMax(&m[(size_t)dst * H + hh], f2key(v));
}

// ---------- edge pass B: z[dst,h] += ex ; num[dst,hc] += ex * h[src,hc] ----------
template<int H, int C>
__global__ void edge_accum_k(const long long* __restrict__ ei, long long Eg, long long E,
                             const float* __restrict__ asrc, const float* __restrict__ adst,
                             const unsigned* __restrict__ m, const float* __restrict__ hfeat,
                             float* __restrict__ z, float* __restrict__ num) {
  const int HC = H * C;
  long long id = (long long)blockIdx.x * blockDim.x + threadIdx.x;
  if (id >= Eg * HC) return;
  long long e = id / HC;
  int hc = (int)(id - e * HC);
  int hh = hc / C;
  int src, dst;
  if (e < E) { src = (int)ei[e]; dst = (int)ei[E + e]; }
  else       { src = dst = (int)(e - E); }
  float v = asrc[(size_t)src * H + hh] + adst[(size_t)dst * H + hh];
  v = v > 0.f ? v : NEG_SLOPE * v;
  float mm = key2f(m[(size_t)dst * H + hh]);
  float ex = __expf(v - mm);
  if (hc == hh * C) atomicAdd(&z[(size_t)dst * H + hh], ex);
  atomicAdd(&num[(size_t)dst * HC + hc], ex * hfeat[(size_t)src * HC + hc]);
}

// ---------- normalize + bias + ELU (writes layer-2 input in place) ----------
template<int H, int C>
__global__ void norm_elu(float* __restrict__ num, const float* __restrict__ z,
                         const float* __restrict__ b, int N) {
  const int HC = H * C;
  long long id = (long long)blockIdx.x * blockDim.x + threadIdx.x;
  if (id >= (long long)N * HC) return;
  long long n = id / HC;
  int hc = (int)(id - n * HC);
  int hh = hc / C;
  float v = num[id] / (z[n * H + hh] + 1e-16f) + b[hc];
  num[id] = v > 0.f ? v : (__expf(v) - 1.f);
}

// ---------- final normalize + bias -> d_out ----------
template<int K>
__global__ void norm_out(const float* __restrict__ num, const float* __restrict__ z,
                         const float* __restrict__ b, float* __restrict__ out, int N) {
  long long id = (long long)blockIdx.x * blockDim.x + threadIdx.x;
  if (id >= (long long)N * K) return;
  long long n = id / K;
  int k = (int)(id - n * K);
  out[id] = num[id] / (z[n] + 1e-16f) + b[k];
}

static inline unsigned nblk(long long tot, int bs) { return (unsigned)((tot + bs - 1) / bs); }

extern "C" void kernel_launch(void* const* d_in, const int* in_sizes, int n_in,
                              void* d_out, int out_size, void* d_ws, size_t ws_size,
                              hipStream_t stream) {
  const float*     x    = (const float*)d_in[0];
  const long long* ei   = (const long long*)d_in[1];   // int64 [2,E]
  const float*     W1   = (const float*)d_in[2];
  const float*     aS1  = (const float*)d_in[3];
  const float*     aD1  = (const float*)d_in[4];
  const float*     b1   = (const float*)d_in[5];
  const float*     W2   = (const float*)d_in[6];
  const float*     aS2  = (const float*)d_in[7];
  const float*     aD2  = (const float*)d_in[8];
  const float*     b2   = (const float*)d_in[9];

  const int H1 = 8, C1 = 8, HC1 = 64, K2 = 40, K2P = 48;
  const int F  = in_sizes[2] / HC1;            // 512
  const int N  = in_sizes[0] / F;              // 100000 (multiple of 16)
  const long long E  = in_sizes[1] / 2;        // 3.2M
  const long long Eg = E + N;                  // + self loops

  // ---- carve workspace ----
  char* w = (char*)d_ws;
  size_t off = 0;
  auto carve = [&](size_t bytes) -> void* {
    void* p = w + off;
    off = (off + bytes + 255) & ~(size_t)255;
    return p;
  };
  bf16*     Wt1  = (bf16*)    carve((size_t)HC1 * F * sizeof(bf16));
  bf16*     Wt2  = (bf16*)    carve((size_t)K2P * HC1 * sizeof(bf16));
  float*    h1   = (float*)   carve((size_t)N * HC1 * sizeof(float));
  float*    as1  = (float*)   carve((size_t)N * H1 * sizeof(float));
  float*    ad1  = (float*)   carve((size_t)N * H1 * sizeof(float));
  unsigned* m1   = (unsigned*)carve((size_t)N * H1 * sizeof(unsigned));
  float*    z1   = (float*)   carve((size_t)N * H1 * sizeof(float));
  float*    num1 = (float*)   carve((size_t)N * HC1 * sizeof(float));  // -> x2 in place
  float*    h2   = (float*)   carve((size_t)N * K2 * sizeof(float));
  float*    as2  = (float*)   carve((size_t)N * sizeof(float));
  float*    ad2  = (float*)   carve((size_t)N * sizeof(float));
  unsigned* m2   = (unsigned*)carve((size_t)N * sizeof(unsigned));
  float*    z2   = (float*)   carve((size_t)N * sizeof(float));
  float*    num2 = (float*)   carve((size_t)N * K2 * sizeof(float));
  (void)ws_size; (void)n_in; (void)out_size;

  // ---- zero accumulators (every call: graph replays must be self-contained) ----
  hipMemsetAsync(m1,   0, (size_t)N * H1 * 4, stream);   // key 0 == "below every real"
  hipMemsetAsync(z1,   0, (size_t)N * H1 * 4, stream);
  hipMemsetAsync(num1, 0, (size_t)N * HC1 * 4, stream);
  hipMemsetAsync(m2,   0, (size_t)N * 4, stream);
  hipMemsetAsync(z2,   0, (size_t)N * 4, stream);
  hipMemsetAsync(num2, 0, (size_t)N * K2 * 4, stream);

  // ---- weight prep (bf16 + transpose) ----
  prep_w<<<nblk((long long)HC1 * F, 256), 256, 0, stream>>>(W1, Wt1, F, HC1, HC1);
  prep_w<<<nblk((long long)K2P * HC1, 256), 256, 0, stream>>>(W2, Wt2, HC1, K2, K2P);

  // ================= layer 1 =================
  {
    int tiles = N / 16;                       // 6250
    wmma_gemm<4><<<nblk(tiles, 4), 128, 0, stream>>>(x, Wt1, h1, N, F, HC1);
  }
  att_scores<8, 8><<<nblk((long long)N * H1, 256), 256, 0, stream>>>(h1, aS1, aD1, as1, ad1, N);
  edge_max_k<8><<<nblk(Eg * H1, 256), 256, 0, stream>>>(ei, Eg, E, as1, ad1, m1);
  edge_accum_k<8, 8><<<nblk(Eg * HC1, 256), 256, 0, stream>>>(ei, Eg, E, as1, ad1, m1, h1, z1, num1);
  norm_elu<8, 8><<<nblk((long long)N * HC1, 256), 256, 0, stream>>>(num1, z1, b1, N);

  // ================= layer 2 =================
  {
    int tiles = N / 16;
    wmma_gemm<3><<<nblk(tiles, 4), 128, 0, stream>>>(num1 /*x2*/, Wt2, h2, N, HC1, K2);
  }
  att_scores<1, 40><<<nblk((long long)N, 256), 256, 0, stream>>>(h2, aS2, aD2, as2, ad2, N);
  edge_max_k<1><<<nblk(Eg, 256), 256, 0, stream>>>(ei, Eg, E, as2, ad2, m2);
  edge_accum_k<1, 40><<<nblk(Eg * K2, 256), 256, 0, stream>>>(ei, Eg, E, as2, ad2, m2, h2, z2, num2);
  norm_out<40><<<nblk((long long)N * K2, 256), 256, 0, stream>>>(num2, z2, b2, (float*)d_out, N);
}
